// CIGA_747324310137
// MI455X (gfx1250) — matrix operand order repaired and checked by reading, not compiled, not run on hardware
//
#include <hip/hip_runtime.h>
#include <hip/hip_bf16.h>

typedef __attribute__((ext_vector_type(16))) _Float16 v16h;
typedef __attribute__((ext_vector_type(8)))  _Float16 v8h;
typedef __attribute__((ext_vector_type(8)))  float    v8f;

#define HID 64
#define NSEG 512
#define SEG_CAP 4096

union FragA { v16h v; v8h q[2]; _Float16 h[16]; };
union AccC  { v8f v; float f[8]; };

__device__ inline void load8_f16(const float* __restrict__ p, _Float16* dst) {
  float4 a = *reinterpret_cast<const float4*>(p);
  float4 b = *reinterpret_cast<const float4*>(p + 4);
  dst[0] = (_Float16)a.x; dst[1] = (_Float16)a.y;
  dst[2] = (_Float16)a.z; dst[3] = (_Float16)a.w;
  dst[4] = (_Float16)b.x; dst[5] = (_Float16)b.y;
  dst[6] = (_Float16)b.z; dst[7] = (_Float16)b.w;
}

// ---------------------------------------------------------------------------
// Kernel 1: WMMA edge scorer.  One wave32 = 32 edges (two 16-row A tiles,
// sharing each B fragment -> halves LDS B traffic per WMMA).
// h[32,256] = relu(edge_rep[32,128] @ W1[128,256] + b1);  att = h @ W2 + b2.
// W1 staged in LDS as f16, transposed, pitch 136 halves (bank-conflict-free).
// ---------------------------------------------------------------------------
__global__ __launch_bounds__(256) void score_kernel(
    const float* __restrict__ emb, const int* __restrict__ ei,
    const int* __restrict__ batch,
    const float* __restrict__ W1, const float* __restrict__ b1,
    const float* __restrict__ W2, const float* __restrict__ b2,
    float* __restrict__ edge_att, int* __restrict__ deg, int nE)
{
  __shared__ __align__(32) _Float16 sW1[256][136];   // [n][k], pitch 136
  __shared__ float sB1[256];
  __shared__ float sW2[256];
  const int tid = threadIdx.x;
  for (int idx = tid; idx < 128 * 256; idx += 256) {
    const int k = idx >> 8, n = idx & 255;           // W1 row-major [128,256]
    sW1[n][k] = (_Float16)W1[idx];
  }
  sB1[tid] = b1[tid];
  sW2[tid] = W2[tid];
  __syncthreads();

  const int wave = tid >> 5;
  const int lane = tid & 31;
  const int hi   = lane >> 4;      // lane group (K/M split per ISA layouts)
  const int rm   = lane & 15;      // A-row / B-col within group
  const int tileBase = (blockIdx.x * 8 + wave) * 32;
  if (tileBase >= nE) return;      // wave-uniform: EXEC stays all-ones for WMMA

  // Gather + convert A fragments for both 16-edge tiles.
  // Lane (hi,rm) holds row rm of tile t, K in [kb,kb+8) and [kb+16,kb+24),
  // kb = kc*32 + hi*8 (ISA 16-bit A 16x32 layout).
  FragA afrag[2][4];
  int nrT[2];
#pragma unroll
  for (int t = 0; t < 2; ++t) {
    const int e  = tileBase + t * 16 + rm;
    const int ec = (e < nE) ? e : (nE - 1);
    const int nr = ei[ec];
    const int nc = ei[nE + ec];
    nrT[t] = nr;
    const float* pr = emb + (long long)nr * HID;
    const float* pc = emb + (long long)nc * HID;
#pragma unroll
    for (int kc = 0; kc < 4; ++kc) {
      const int c0 = kc * 32 + hi * 8;
      const int c1 = c0 + 16;
      const float* p0 = (c0 < HID) ? (pr + c0) : (pc + (c0 - HID));
      const float* p1 = (c1 < HID) ? (pr + c1) : (pc + (c1 - HID));
      load8_f16(p0, &afrag[t][kc].h[0]);
      load8_f16(p1, &afrag[t][kc].h[8]);
    }
  }

  float partial[2][8];
#pragma unroll
  for (int t = 0; t < 2; ++t)
#pragma unroll
    for (int r = 0; r < 8; ++r) partial[t][r] = 0.f;

  for (int nt = 0; nt < 16; ++nt) {
    const int n = nt * 16 + rm;                      // output column
    AccC acc0{}, acc1{};
#pragma unroll
    for (int kc = 0; kc < 4; ++kc) {
      // B 32x16: lane holds col n, K in [kb,kb+8) and [kb+16,kb+24), kb = kc*32+hi*8
      const int kb = kc * 32 + hi * 8;
      FragA bfrag;
      bfrag.q[0] = *reinterpret_cast<const v8h*>(&sW1[n][kb]);
      bfrag.q[1] = *reinterpret_cast<const v8h*>(&sW1[n][kb + 16]);
      acc0.v = __builtin_amdgcn_wmma_f32_16x16x32_f16(
          false, afrag[0][kc].v, false, bfrag.v, (short)0, acc0.v, false, false);
      acc1.v = __builtin_amdgcn_wmma_f32_16x16x32_f16(
          false, afrag[1][kc].v, false, bfrag.v, (short)0, acc1.v, false, false);
    }
    const float bv = sB1[n];
    const float wv = sW2[n];
#pragma unroll
    for (int r = 0; r < 8; ++r) {                    // C: row = hi*8+r, col = n
      float h0 = fmaxf(acc0.f[r] + bv, 0.f);
      float h1 = fmaxf(acc1.f[r] + bv, 0.f);
      partial[0][r] += h0 * wv;                      // fuse h @ W2
      partial[1][r] += h1 * wv;
    }
  }
  // reduce over the 16 column-lanes within each half-wave
#pragma unroll
  for (int t = 0; t < 2; ++t)
#pragma unroll
    for (int r = 0; r < 8; ++r) {
      float v = partial[t][r];
      v += __shfl_xor(v, 1, 32);
      v += __shfl_xor(v, 2, 32);
      v += __shfl_xor(v, 4, 32);
      v += __shfl_xor(v, 8, 32);
      partial[t][r] = v;
    }
  const float b2v = b2[0];
  if (rm == 0) {
#pragma unroll
    for (int t = 0; t < 2; ++t)
#pragma unroll
      for (int r = 0; r < 8; ++r) {
        const int ew = tileBase + t * 16 + hi * 8 + r;
        if (ew < nE) edge_att[ew] = partial[t][r] + b2v;
      }
  }
  // seg histogram: each lane owns one edge (tile hi, row rm)
  {
    const int ea = tileBase + hi * 16 + rm;
    if (ea < nE) atomicAdd(&deg[batch[nrT[hi]]], 1);
  }
}

// ---------------------------------------------------------------------------
// Zero deg + node flags
// ---------------------------------------------------------------------------
__global__ void zero3_kernel(int* __restrict__ deg, int nDeg,
                             int* __restrict__ cflag, int* __restrict__ sflag,
                             int nNodes) {
  const int i = blockIdx.x * blockDim.x + threadIdx.x;
  if (i < nDeg) deg[i] = 0;
  if (i < nNodes) { cflag[i] = 0; sflag[i] = 0; }
}

// ---------------------------------------------------------------------------
// Scan segment degrees (512, one block); k = ceil(0.5*deg); reset fill ctrs
// ---------------------------------------------------------------------------
__global__ void segscan_kernel(const int* __restrict__ deg,
                               int* __restrict__ seg_start,
                               int* __restrict__ kk,
                               int* __restrict__ seg_fill, int nSeg) {
  __shared__ int tmp[NSEG];
  const int t = threadIdx.x;
  const int d = (t < nSeg) ? deg[t] : 0;
  tmp[t] = d;
  __syncthreads();
  for (int off = 1; off < NSEG; off <<= 1) {
    const int add = (t >= off) ? tmp[t - off] : 0;
    __syncthreads();
    tmp[t] += add;
    __syncthreads();
  }
  if (t < nSeg) {
    seg_start[t] = tmp[t] - d;       // exclusive prefix
    kk[t] = (d + 1) >> 1;            // ceil(0.5 * deg)
    seg_fill[t] = 0;
  }
}

// ---------------------------------------------------------------------------
// Bucket edges by segment (order within segment irrelevant: final keep mask
// depends only on the (value, edge-id) multiset -> deterministic outputs)
// ---------------------------------------------------------------------------
__global__ void scatter_kernel(const float* __restrict__ edge_att,
                               const int* __restrict__ ei,
                               const int* __restrict__ batch,
                               const int* __restrict__ seg_start,
                               int* __restrict__ seg_fill,
                               float* __restrict__ gval, int* __restrict__ gid,
                               int nE) {
  const int e = blockIdx.x * blockDim.x + threadIdx.x;
  if (e >= nE) return;
  const int s = batch[ei[e]];
  const int p = atomicAdd(&seg_fill[s], 1);
  const int dst = seg_start[s] + p;
  gval[dst] = edge_att[e];
  gid[dst]  = e;
}

// ---------------------------------------------------------------------------
// Per-segment top-k via in-LDS bitonic sort on key (val desc, id asc):
// identical order to the reference's stable descending argsort (incl. ties).
// ---------------------------------------------------------------------------
__global__ __launch_bounds__(256) void topk_kernel(
    const float* __restrict__ gval, const int* __restrict__ gid,
    const int* __restrict__ seg_start, const int* __restrict__ deg,
    const int* __restrict__ kk, int* __restrict__ keep) {
  const int s = blockIdx.x;
  const int n = deg[s];
  const int base = seg_start[s];
  const int klim = kk[s];
  if (n <= 0) return;
  __shared__ float sv[SEG_CAP];
  __shared__ int   sid[SEG_CAP];
  if (n <= SEG_CAP) {
    int m = 1;
    while (m < n) m <<= 1;                     // next power of two
    for (int i = threadIdx.x; i < m; i += blockDim.x) {
      if (i < n) { sv[i] = gval[base + i]; sid[i] = gid[base + i]; }
      else       { sv[i] = -__builtin_huge_valf(); sid[i] = 0x7fffffff; }
    }
    __syncthreads();
    for (int ksz = 2; ksz <= m; ksz <<= 1) {
      for (int j = ksz >> 1; j > 0; j >>= 1) {
        for (int i = threadIdx.x; i < m; i += blockDim.x) {
          const int ixj = i ^ j;
          if (ixj > i) {
            const float va = sv[i],  vb = sv[ixj];
            const int   ia = sid[i], ib = sid[ixj];
            const bool up = ((i & ksz) == 0);
            // a_first: a precedes b in the desired final order
            const bool a_first = (va > vb) || (va == vb && ia < ib);
            const bool sw = up ? !a_first : a_first;
            if (sw) { sv[i] = vb; sid[i] = ib; sv[ixj] = va; sid[ixj] = ia; }
          }
        }
        __syncthreads();
      }
    }
    for (int p = threadIdx.x; p < n; p += blockDim.x)
      keep[sid[p]] = (p < klim) ? 1 : 0;
  } else {
    // fallback (statistically unreachable): exact O(n^2) rank from global
    for (int i = threadIdx.x; i < n; i += blockDim.x) {
      const float vi = gval[base + i];
      const int  idi = gid[base + i];
      int rank = 0;
      for (int j = 0; j < n; ++j) {
        const float vj = gval[base + j];
        const int  idj = gid[base + j];
        rank += (vj > vi) || (vj == vi && idj < idi);
      }
      keep[idi] = (rank < klim) ? 1 : 0;
    }
  }
}

// ---------------------------------------------------------------------------
// scores rows + endpoint flags for relabeling
// ---------------------------------------------------------------------------
__global__ void finalize_kernel(const float* __restrict__ edge_att,
                                const int* __restrict__ keep,
                                const int* __restrict__ ei,
                                float* __restrict__ out_scores,
                                int* __restrict__ cflag, int* __restrict__ sflag,
                                int nE) {
  const int e = blockIdx.x * blockDim.x + threadIdx.x;
  if (e >= nE) return;
  const float f = edge_att[e];
  const int kp = keep[e];
  out_scores[e]                  = f;
  out_scores[(size_t)nE + e]     = kp ? f : 0.f;
  out_scores[(size_t)2 * nE + e] = kp ? 0.f : -f;
  const int r = ei[e], c = ei[nE + e];
  if (kp) { cflag[r] = 1; cflag[c] = 1; }
  else    { sflag[r] = 1; sflag[c] = 1; }
}

// ---------------------------------------------------------------------------
// Single-block chunked inclusive scan (wave32 shfl scans, 4 barriers/chunk):
// idx[v] = flag ? cumsum-1 : -1
// ---------------------------------------------------------------------------
__global__ __launch_bounds__(1024) void nodescan_kernel(
    const int* __restrict__ flag, int* __restrict__ idx, int n) {
  __shared__ int wsum[32];
  __shared__ int carryS;
  const int t = threadIdx.x;
  const int lane = t & 31, wid = t >> 5;
  if (t == 0) carryS = 0;
  __syncthreads();
  for (int base = 0; base < n; base += 1024) {
    const int i = base + t;
    const int f = (i < n) ? flag[i] : 0;
    int v = f;                                   // wave inclusive scan
#pragma unroll
    for (int off = 1; off < 32; off <<= 1) {
      const int u = __shfl_up(v, off, 32);
      if (lane >= off) v += u;
    }
    if (lane == 31) wsum[wid] = v;
    __syncthreads();
    if (wid == 0) {                              // scan the 32 wave totals
      int w = wsum[lane];
#pragma unroll
      for (int off = 1; off < 32; off <<= 1) {
        const int u = __shfl_up(w, off, 32);
        if (lane >= off) w += u;
      }
      wsum[lane] = w;
    }
    __syncthreads();
    const int waveOff = (wid == 0) ? 0 : wsum[wid - 1];
    const int carry = carryS;
    if (i < n) idx[i] = f ? (carry + waveOff + v - 1) : -1;
    __syncthreads();
    if (t == 0) carryS += wsum[31];
    __syncthreads();
  }
}

// ---------------------------------------------------------------------------
// Relabeled edge indices (-1 for dropped edges)
// ---------------------------------------------------------------------------
__global__ void relabel_kernel(const int* __restrict__ ei,
                               const int* __restrict__ keep,
                               const int* __restrict__ cidx,
                               const int* __restrict__ sidx,
                               int* __restrict__ out_c, int* __restrict__ out_s,
                               int nE) {
  const int e = blockIdx.x * blockDim.x + threadIdx.x;
  if (e >= nE) return;
  const int kp = keep[e];
  const int r = ei[e], c = ei[nE + e];
  out_c[e]              = kp ? cidx[r] : -1;
  out_c[(size_t)nE + e] = kp ? cidx[c] : -1;
  out_s[e]              = kp ? -1 : sidx[r];
  out_s[(size_t)nE + e] = kp ? -1 : sidx[c];
}

// ---------------------------------------------------------------------------
extern "C" void kernel_launch(void* const* d_in, const int* in_sizes, int n_in,
                              void* d_out, int out_size, void* d_ws, size_t ws_size,
                              hipStream_t stream) {
  const float* emb   = (const float*)d_in[0];
  const int*   ei    = (const int*)d_in[1];
  const int*   batch = (const int*)d_in[2];
  const float* W1    = (const float*)d_in[3];
  const float* b1    = (const float*)d_in[4];
  const float* W2    = (const float*)d_in[5];
  const float* b2    = (const float*)d_in[6];
  const int nE     = in_sizes[1] / 2;
  const int nNodes = in_sizes[2];
  const int nSeg   = NSEG;

  char* ws = (char*)d_ws;
  auto alloc = [&](size_t bytes) {
    char* p = ws;
    ws += (bytes + 255) & ~(size_t)255;
    return p;
  };
  float* edge_att  = (float*)alloc((size_t)nE * 4);
  float* gval      = (float*)alloc((size_t)nE * 4);
  int*   gid       = (int*)  alloc((size_t)nE * 4);
  int*   keep      = (int*)  alloc((size_t)nE * 4);
  int*   deg       = (int*)  alloc((size_t)nSeg * 4);
  int*   seg_start = (int*)  alloc((size_t)nSeg * 4);
  int*   kk        = (int*)  alloc((size_t)nSeg * 4);
  int*   seg_fill  = (int*)  alloc((size_t)nSeg * 4);
  int*   cflag     = (int*)  alloc((size_t)nNodes * 4);
  int*   sflag     = (int*)  alloc((size_t)nNodes * 4);
  int*   cidx      = (int*)  alloc((size_t)nNodes * 4);
  int*   sidx      = (int*)  alloc((size_t)nNodes * 4);

  float* out_scores = (float*)d_out;                  // [3, E] f32
  int*   out_c      = (int*)d_out + (size_t)3 * nE;   // [2, E] i32 bits
  int*   out_s      = (int*)d_out + (size_t)5 * nE;   // [2, E] i32 bits

  zero3_kernel<<<(nNodes + 255) / 256, 256, 0, stream>>>(deg, nSeg, cflag, sflag, nNodes);
  score_kernel<<<(nE + 255) / 256, 256, 0, stream>>>(emb, ei, batch, W1, b1, W2, b2,
                                                     edge_att, deg, nE);
  segscan_kernel<<<1, NSEG, 0, stream>>>(deg, seg_start, kk, seg_fill, nSeg);
  scatter_kernel<<<(nE + 255) / 256, 256, 0, stream>>>(edge_att, ei, batch, seg_start,
                                                       seg_fill, gval, gid, nE);
  topk_kernel<<<nSeg, 256, 0, stream>>>(gval, gid, seg_start, deg, kk, keep);
  finalize_kernel<<<(nE + 255) / 256, 256, 0, stream>>>(edge_att, keep, ei, out_scores,
                                                        cflag, sflag, nE);
  nodescan_kernel<<<1, 1024, 0, stream>>>(cflag, cidx, nNodes);
  nodescan_kernel<<<1, 1024, 0, stream>>>(sflag, sidx, nNodes);
  relabel_kernel<<<(nE + 255) / 256, 256, 0, stream>>>(ei, keep, cidx, sidx, out_c, out_s, nE);
}
